// BiRNN_18622978196239
// MI455X (gfx1250) — matrix operand order repaired and checked by reading, not compiled, not run on hardware
//
#include <hip/hip_runtime.h>

// ---------------------------------------------------------------------------
// BiRNN (2-layer, bidirectional, tanh) for MI455X / gfx1250, bf16 WMMA path.
// B=128, S=512, E=512, H=1024, V=50000, C=2
// ---------------------------------------------------------------------------

#define BB 128
#define SS 512
#define EE 512
#define HH 1024
#define CC 2
#define NT 2            // ntiles per block (B-slab staged in LDS, A reused NT x)

typedef __attribute__((ext_vector_type(16))) __bf16 v16bf;
typedef __attribute__((ext_vector_type(8)))  __bf16 bf16x8;
typedef __attribute__((ext_vector_type(8)))  float  v8f;
typedef __attribute__((ext_vector_type(4)))  int    v4i;

#if __has_builtin(__builtin_amdgcn_global_load_async_to_lds_b128) && \
    __has_builtin(__builtin_amdgcn_s_wait_asynccnt)
#define HAVE_ASYNC_LDS 1
typedef __attribute__((address_space(1))) v4i g_v4i;   // global int4
typedef __attribute__((address_space(3))) v4i l_v4i;   // LDS int4
#else
#define HAVE_ASYNC_LDS 0
#endif

__device__ __forceinline__ __bf16 f2bf(float f) {
    unsigned u = __builtin_bit_cast(unsigned, f);
    unsigned r = (u + 0x7FFFu + ((u >> 16) & 1u)) >> 16;
    unsigned short s = (unsigned short)r;
    return __builtin_bit_cast(__bf16, s);
}
__device__ __forceinline__ float bf2f(__bf16 h) {
    unsigned short s = __builtin_bit_cast(unsigned short, h);
    unsigned u = ((unsigned)s) << 16;
    return __builtin_bit_cast(float, u);
}

// ---------------------------------------------------------------------------
// Embedding gather: e_bf[(s*B+b)*E + j] = bf16(emb[x[b*S+s]*E + j])
// ---------------------------------------------------------------------------
__global__ __launch_bounds__(256) void gather_embed_kernel(
    const int* __restrict__ x, const float* __restrict__ emb,
    __bf16* __restrict__ e_bf)
{
    size_t idx = (size_t)blockIdx.x * 256 + threadIdx.x;   // over S*B*E
    int j   = (int)(idx % EE);
    int row = (int)(idx / EE);          // row = s*B + b
    int s = row >> 7;                   // / B
    int b = row & (BB - 1);
    int token = x[(size_t)b * SS + s];
    e_bf[idx] = f2bf(emb[(size_t)token * EE + j]);
}

// ---------------------------------------------------------------------------
// Weight transpose + convert: WT[n*K + k] = bf16(W[k*N + n])   (W is K x N)
// ---------------------------------------------------------------------------
__global__ __launch_bounds__(256) void transpose_bf16_kernel(
    const float* __restrict__ W, __bf16* __restrict__ WT, int K, int N)
{
    size_t idx = (size_t)blockIdx.x * 256 + threadIdx.x;
    if (idx >= (size_t)K * N) return;
    int n = (int)(idx % N);
    int k = (int)(idx / N);
    WT[(size_t)n * K + k] = f2bf(W[idx]);
}

__global__ __launch_bounds__(256) void bias_sum_kernel(
    const float* __restrict__ bi, const float* __restrict__ bh,
    float* __restrict__ bs)
{
    int n = blockIdx.x * 256 + threadIdx.x;
    if (n < HH) bs[n] = bi[n] + bh[n];
}

__global__ __launch_bounds__(256) void zero_bf16_kernel(__bf16* p, int count)
{
    int i = blockIdx.x * 256 + threadIdx.x;
    if (i < count) p[i] = f2bf(0.0f);
}

// ---------------------------------------------------------------------------
// Unified WMMA tile kernel.
//   Block = 256 threads = 8 waves. Block owns an M-slab of 128 rows
//   (wave w -> rows [mslab*128 + w*16, +16)) and an N-slab of NT*16 columns.
//   The W^T slab (NT*16 rows x K) is staged into LDS once (async-to-LDS if
//   available), then every wave reuses its A fragment for NT WMMAs whose B
//   fragments come from LDS.
//
//   Epilogue (runtime-uniform):
//     bias != 0 : add bias[n]                         (Xi projection GEMM)
//     xi   != 0 : val = tanh(val + xi[m,n])           (recurrent step)
//     ys   != 0 : also store to ys[m,n]
// ---------------------------------------------------------------------------
__global__ __launch_bounds__(256) void wmma_tile_kernel(
    const __bf16* __restrict__ A,     // M x K bf16 (row remap if rev)
    const __bf16* __restrict__ WT,    // N x K bf16 (WT[n*K+k] = W[k,n])
    const float*  __restrict__ bias,  // N fp32 or nullptr
    const __bf16* __restrict__ xi,    // M x N bf16 or nullptr
    __bf16* __restrict__ Cout,        // M x N bf16
    __bf16* __restrict__ ys,          // M x N bf16 or nullptr
    int N, int K, int nslabs, int rev)
{
    extern __shared__ __bf16 Bs[];    // NT*16 rows x K

    const int w    = threadIdx.x >> 5;
    const int lane = threadIdx.x & 31;
    const int r    = lane & 15;
    const int hi   = lane >> 4;

    const int nslab = blockIdx.x % nslabs;
    const int mslab = blockIdx.x / nslabs;
    const int n0    = nslab * (NT * 16);          // first column of N-slab

    // ---- stage W^T slab into LDS ---------------------------------------
    {
        const __bf16* bsrc = WT + (size_t)n0 * K;
        const int chunks = (NT * 16 * K) / 8;     // 16B chunks
#if HAVE_ASYNC_LDS
        for (int c = threadIdx.x; c < chunks; c += 256) {
            __builtin_amdgcn_global_load_async_to_lds_b128(
                (g_v4i*)(bsrc + (size_t)c * 8),
                (l_v4i*)(Bs + (size_t)c * 8),
                0, 0);
        }
        __builtin_amdgcn_s_wait_asynccnt(0);
        __syncthreads();
#else
        for (int c = threadIdx.x; c < chunks; c += 256)
            *(bf16x8*)(Bs + (size_t)c * 8) = *(const bf16x8*)(bsrc + (size_t)c * 8);
        __syncthreads();
#endif
    }

    // ---- per-lane A row pointer (with optional sequence reversal) -------
    const int row = mslab * 128 + w * 16 + r;
    int srow = row;
    if (rev) {                       // read e[S-1-s] for the backward scan
        int s = row >> 7;            // / B
        int b = row & (BB - 1);
        srow = (SS - 1 - s) * BB + b;
    }
    const __bf16* aptr = A + (size_t)srow * K;
    const __bf16* bls  = Bs + (size_t)r * K;      // lane's B rows (n-local = r)

    v8f acc[NT];
    #pragma unroll
    for (int q = 0; q < NT; ++q) { v8f z = {}; acc[q] = z; }

    for (int k0 = 0; k0 < K; k0 += 32) {
        // A 16x32 bf16 fragment, ISA lane layout:
        //   lanes 0-15 : K = [0..7] then [16..23]; lanes 16-31: +8
        bf16x8 alo = *(const bf16x8*)(aptr + k0 + hi * 8);
        bf16x8 ahi = *(const bf16x8*)(aptr + k0 + 16 + hi * 8);
        v16bf a = __builtin_shufflevector(alo, ahi,
                    0,1,2,3,4,5,6,7,8,9,10,11,12,13,14,15);
        #pragma unroll
        for (int q = 0; q < NT; ++q) {
            // B 32x16 fragment from LDS: K = [0..7] then [16..23]; lanes 16-31: +8
            bf16x8 blo = *(const bf16x8*)(bls + (size_t)q * 16 * K + k0 + hi * 8);
            bf16x8 bhi = *(const bf16x8*)(bls + (size_t)q * 16 * K + k0 + 16 + hi * 8);
            v16bf b = __builtin_shufflevector(blo, bhi,
                    0,1,2,3,4,5,6,7,8,9,10,11,12,13,14,15);
            acc[q] = __builtin_amdgcn_wmma_f32_16x16x32_bf16(
                         false, a, false, b, (short)0, acc[q], false, false);
        }
    }

    // ---- epilogue -------------------------------------------------------
    const int m0 = mslab * 128 + w * 16 + hi * 8;
    #pragma unroll
    for (int q = 0; q < NT; ++q) {
        const int n = n0 + q * 16 + r;
        const float bn = bias ? bias[n] : 0.0f;
        #pragma unroll
        for (int v = 0; v < 8; ++v) {
            const int m = m0 + v;
            float val = acc[q][v] + bn;
            if (xi) val = tanhf(val + bf2f(xi[(size_t)m * N + n]));
            __bf16 bv = f2bf(val);
            Cout[(size_t)m * N + n] = bv;
            if (ys) ys[(size_t)m * N + n] = bv;
        }
    }
}

// ---------------------------------------------------------------------------
// Head: out = (concat(hf,hb) @ fc1_W + fc1_b) @ fc2_W + fc2_b
// Grid: B blocks x 64 threads.
// ---------------------------------------------------------------------------
__global__ __launch_bounds__(64) void head_kernel(
    const __bf16* __restrict__ hf, const __bf16* __restrict__ hb,
    const float* __restrict__ fc1W, const float* __restrict__ fc1b,
    const float* __restrict__ fc2W, const float* __restrict__ fc2b,
    float* __restrict__ out)
{
    __shared__ float z[64];
    int b = blockIdx.x;
    int j = threadIdx.x;
    float acc = fc1b[j];
    for (int k = 0; k < HH; ++k)
        acc += bf2f(hf[(size_t)b * HH + k]) * fc1W[(size_t)k * 64 + j];
    for (int k = 0; k < HH; ++k)
        acc += bf2f(hb[(size_t)b * HH + k]) * fc1W[(size_t)(HH + k) * 64 + j];
    z[j] = acc;
    __syncthreads();
    if (j < CC) {
        float o = fc2b[j];
        for (int q = 0; q < 64; ++q) o += z[q] * fc2W[q * CC + j];
        out[b * CC + j] = o;
    }
}

// ---------------------------------------------------------------------------
// Host orchestration
// ---------------------------------------------------------------------------
extern "C" void kernel_launch(void* const* d_in, const int* in_sizes, int n_in,
                              void* d_out, int out_size, void* d_ws, size_t ws_size,
                              hipStream_t stream)
{
    (void)in_sizes; (void)n_in; (void)out_size; (void)ws_size;

    const int*   x    = (const int*)  d_in[0];
    const float* emb  = (const float*)d_in[1];
    // per-layer weights: (Wi, bi, Wh, bh) for f0,f1,b0,b1
    const float* Wi[4] = { (const float*)d_in[2],  (const float*)d_in[6],
                           (const float*)d_in[10], (const float*)d_in[14] };
    const float* bi[4] = { (const float*)d_in[3],  (const float*)d_in[7],
                           (const float*)d_in[11], (const float*)d_in[15] };
    const float* Wh[4] = { (const float*)d_in[4],  (const float*)d_in[8],
                           (const float*)d_in[12], (const float*)d_in[16] };
    const float* bh[4] = { (const float*)d_in[5],  (const float*)d_in[9],
                           (const float*)d_in[13], (const float*)d_in[17] };
    const int Kin[4] = { EE, HH, EE, HH };   // input dim of each layer
    const float* fc1W = (const float*)d_in[18];
    const float* fc1b = (const float*)d_in[19];
    const float* fc2W = (const float*)d_in[20];
    const float* fc2b = (const float*)d_in[21];
    float* out = (float*)d_out;

    // ---- workspace carve-out -------------------------------------------
    char* ws = (char*)d_ws;
    size_t off = 0;
    auto carve = [&](size_t bytes) -> char* {
        off = (off + 255) & ~(size_t)255;
        char* p = ws + off;
        off += bytes;
        return p;
    };
    __bf16* e_bf = (__bf16*)carve((size_t)SS * BB * EE * 2);   //  64 MB
    __bf16* Xi   = (__bf16*)carve((size_t)SS * BB * HH * 2);   // 128 MB
    __bf16* ys   = (__bf16*)carve((size_t)SS * BB * HH * 2);   // 128 MB
    __bf16* WiT[4]; __bf16* WhT[4]; float* bsum[4];
    for (int l = 0; l < 4; ++l) {
        WiT[l]  = (__bf16*)carve((size_t)HH * Kin[l] * 2);
        WhT[l]  = (__bf16*)carve((size_t)HH * HH * 2);
        bsum[l] = (float*) carve((size_t)HH * 4);
    }
    __bf16* hping[2][2];   // [dir][buf]
    for (int d = 0; d < 2; ++d)
        for (int q = 0; q < 2; ++q)
            hping[d][q] = (__bf16*)carve((size_t)BB * HH * 2);

    // ---- weight prep ----------------------------------------------------
    for (int l = 0; l < 4; ++l) {
        size_t nWi = (size_t)Kin[l] * HH;
        transpose_bf16_kernel<<<(unsigned)((nWi + 255) / 256), 256, 0, stream>>>(
            Wi[l], WiT[l], Kin[l], HH);
        size_t nWh = (size_t)HH * HH;
        transpose_bf16_kernel<<<(unsigned)((nWh + 255) / 256), 256, 0, stream>>>(
            Wh[l], WhT[l], HH, HH);
        bias_sum_kernel<<<(HH + 255) / 256, 256, 0, stream>>>(bi[l], bh[l], bsum[l]);
    }

    // ---- embedding gather ----------------------------------------------
    {
        size_t n = (size_t)SS * BB * EE;
        gather_embed_kernel<<<(unsigned)(n / 256), 256, 0, stream>>>(x, emb, e_bf);
    }

    const int M       = SS * BB;                 // 65536 rows
    const int nslabs  = HH / (16 * NT);          // 32
    const int mslabsG = M / 128;                 // 512
    const int hcount  = BB * HH;

    __bf16* hfinal[2] = { nullptr, nullptr };

    for (int dir = 0; dir < 2; ++dir) {          // 0 = fwd (f0,f1), 1 = bwd (b0,b1)
        int l0 = dir * 2, l1 = dir * 2 + 1;

        // layer 0: Xi = e (maybe reversed) @ Wi0 + (bi0+bh0)
        wmma_tile_kernel<<<(unsigned)(mslabsG * nslabs), 256,
                           NT * 16 * EE * 2, stream>>>(
            e_bf, WiT[l0], bsum[l0], nullptr, Xi, nullptr,
            HH, EE, nslabs, /*rev=*/dir);

        __bf16* prev = hping[dir][0];
        __bf16* next = hping[dir][1];
        zero_bf16_kernel<<<(hcount + 255) / 256, 256, 0, stream>>>(prev, hcount);
        for (int t = 0; t < SS; ++t) {
            wmma_tile_kernel<<<nslabs, 256, NT * 16 * HH * 2, stream>>>(
                prev, WhT[l0], nullptr, Xi + (size_t)t * BB * HH,
                next, ys + (size_t)t * BB * HH,
                HH, HH, nslabs, /*rev=*/0);
            __bf16* tmp = prev; prev = next; next = tmp;
        }

        // layer 1: Xi = ys @ Wi1 + (bi1+bh1)
        wmma_tile_kernel<<<(unsigned)(mslabsG * nslabs), 256,
                           NT * 16 * HH * 2, stream>>>(
            ys, WiT[l1], bsum[l1], nullptr, Xi, nullptr,
            HH, HH, nslabs, /*rev=*/0);

        zero_bf16_kernel<<<(hcount + 255) / 256, 256, 0, stream>>>(prev, hcount);
        for (int t = 0; t < SS; ++t) {
            wmma_tile_kernel<<<nslabs, 256, NT * 16 * HH * 2, stream>>>(
                prev, WhT[l1], nullptr, Xi + (size_t)t * BB * HH,
                next, nullptr,
                HH, HH, nslabs, /*rev=*/0);
            __bf16* tmp = prev; prev = next; next = tmp;
        }
        hfinal[dir] = prev;                      // last written hidden state
    }

    head_kernel<<<BB, 64, 0, stream>>>(hfinal[0], hfinal[1],
                                       fc1W, fc1b, fc2W, fc2b, out);
}
